// FlipFlop_24051816858093
// MI455X (gfx1250) — matrix operand order repaired and checked by reading, not compiled, not run on hardware
//
#include <hip/hip_runtime.h>
#include <hip/hip_bf16.h>
#include <cstdint>

typedef __attribute__((ext_vector_type(16))) _Float16 v16h;
typedef __attribute__((ext_vector_type(8)))  _Float16 v8h;
typedef __attribute__((ext_vector_type(8)))  float    v8f;

#define RNN_B 256
#define RNN_T 1024
#define RNN_I 3
#define RNN_H 128
#define RNN_O 3

#define MTILE  16    // batch rows per workgroup (WMMA M)
#define TCHUNK 128   // timesteps of input staged in LDS per refill
#define HSTR   136   // padded LDS row stride in halves (272B) -> bank-conflict-free A loads

// tanh(x) = 1 - 2/(exp(2x)+1); uses v_exp_f32 + v_rcp_f32, saturates to +/-1 correctly.
__device__ __forceinline__ float fast_tanhf(float x) {
    const float e = __builtin_amdgcn_exp2f(x * 2.8853900817779268f); // exp(2x)
    return 1.0f - 2.0f * __builtin_amdgcn_rcpf(e + 1.0f);
}

__global__ __launch_bounds__(256, 1)
void rnn_scan_kernel(const float* __restrict__ inputs,  // [B,T,I]
                     const float* __restrict__ W_ih,    // [H,I]
                     const float* __restrict__ W_hh,    // [H,H]
                     const float* __restrict__ b_ih,    // [H]
                     const float* __restrict__ b_hh,    // [H]
                     const float* __restrict__ h0,      // [1,H]
                     float* __restrict__ hid)           // [B,T,H]
{
    __shared__ __align__(16) _Float16 lds_h[2][MTILE * HSTR];        // h(t) ping-pong, f16
    __shared__ __align__(16) float    lds_x[TCHUNK * MTILE * RNN_I]; // staged inputs

    const int tid  = threadIdx.x;
    const int lane = tid & 31;
    const int wave = tid >> 5;
    const int hi16 = (lane >= 16) ? 1 : 0;
    const int nloc = lane & 15;
    const int n    = wave * 16 + nloc;   // hidden column owned by this lane (C-layout N)
    const int mhi  = hi16 ? 8 : 0;       // C-layout row offset (VGPR r -> M = r + mhi)
    const int b0   = blockIdx.x * MTILE; // batch tile base

    // ---- resident B fragments: W_hh^T, K=128 in 4 chunks of 32 ------------
    // 16-bit B 32x16 layout: element e = B[k][nloc], k = c*32 + (e<8 ? bkoff+e : 16+bkoff+e-8), bkoff = lane<16 ? 0 : 8
    const int bkoff = hi16 ? 8 : 0;
    v16h bfrag[4];
#pragma unroll
    for (int c = 0; c < 4; ++c) {
        const float* wp = W_hh + (size_t)n * RNN_H + c * 32 + bkoff;
#pragma unroll
        for (int e = 0; e < 16; ++e) bfrag[c][e] = (_Float16)wp[e < 8 ? e : e + 8];
    }
    const float wih0 = W_ih[n * RNN_I + 0];
    const float wih1 = W_ih[n * RNN_I + 1];
    const float wih2 = W_ih[n * RNN_I + 2];
    const float bias = b_ih[n] + b_hh[n];

    // ---- h(-1) = h0 broadcast into buffer 0 -------------------------------
    for (int idx = tid; idx < MTILE * RNN_H; idx += 256) {
        const int r = idx >> 7, c = idx & 127;
        lds_h[0][r * HSTR + c] = (_Float16)h0[c];
    }
    __syncthreads();

    // A-fragment addressing (16-bit A 16x32 layout)
    const int arow  = nloc;          // lanes 0-15 and 16-31 both map to rows 0..15
    const int akoff = hi16 ? 8 : 0;  // lanes 16-31 hold K 8-15 / 24-31

    int cur = 0;
#pragma unroll 1
    for (int t = 0; t < RNN_T; ++t) {
        // ---- refill input chunk every TCHUNK steps ------------------------
        if ((t & (TCHUNK - 1)) == 0) {
            for (int idx = tid; idx < MTILE * TCHUNK * RNN_I; idx += 256) {
                const int b   = idx / (TCHUNK * RNN_I);
                const int rem = idx - b * (TCHUNK * RNN_I);
                const int tl  = rem / RNN_I;
                const int i   = rem - tl * RNN_I;
                lds_x[tl * (MTILE * RNN_I) + b * RNN_I + i] =
                    inputs[((size_t)(b0 + b) * RNN_T + (t + tl)) * RNN_I + i];
            }
            if ((t + TCHUNK < RNN_T) && tid < MTILE) {
                __builtin_prefetch(
                    &inputs[((size_t)(b0 + tid) * RNN_T + (t + TCHUNK)) * RNN_I], 0, 1);
            }
            __syncthreads();
        }

        // ---- fused input projection -> WMMA C operand ---------------------
        const float* xt = &lds_x[(t & (TCHUNK - 1)) * (MTILE * RNN_I)];
        v8f acc;
#pragma unroll
        for (int r = 0; r < 8; ++r) {
            const int m = r + mhi;
            acc[r] = fmaf(xt[m * RNN_I + 0], wih0,
                     fmaf(xt[m * RNN_I + 1], wih1,
                     fmaf(xt[m * RNN_I + 2], wih2, bias)));
        }

        // ---- h(t-1) @ W_hh^T : 4 chained v_wmma_f32_16x16x32_f16 ----------
        const _Float16* hb = lds_h[cur];
#pragma unroll
        for (int c = 0; c < 4; ++c) {
            const v8h lo = *(const v8h*)&hb[arow * HSTR + c * 32 + akoff];
            const v8h hg = *(const v8h*)&hb[arow * HSTR + c * 32 + 16 + akoff];
            const v16h a = __builtin_shufflevector(
                lo, hg, 0, 1, 2, 3, 4, 5, 6, 7, 8, 9, 10, 11, 12, 13, 14, 15);
            acc = __builtin_amdgcn_wmma_f32_16x16x32_f16(
                false, a, false, bfrag[c], (short)0, acc, false, false);
        }

        // ---- tanh, publish h(t) to LDS (f16) and HBM (f32) ----------------
        _Float16* hn = lds_h[cur ^ 1];
#pragma unroll
        for (int r = 0; r < 8; ++r) {
            const float h = fast_tanhf(acc[r]);
            const int m = r + mhi;
            hn[m * HSTR + n] = (_Float16)h;
            hid[((size_t)(b0 + m) * RNN_T + t) * RNN_H + n] = h;
        }
        cur ^= 1;
        __syncthreads();
    }
}

// One wave per (b,t) row: 3 length-128 dot products, wave32 shfl_xor reduction.
__global__ __launch_bounds__(256, 1)
void out_proj_kernel(const float* __restrict__ hid,    // [B,T,H]
                     const float* __restrict__ W_out,  // [O,H]
                     const float* __restrict__ b_out,  // [O]
                     float* __restrict__ out)          // [B,T,O]
{
    const int gwave = (int)((blockIdx.x * blockDim.x + threadIdx.x) >> 5); // b*T + t
    const int lane  = threadIdx.x & 31;

    const float4 h = *(const float4*)(hid + (size_t)gwave * RNN_H + lane * 4);
    float p0, p1, p2;
    {
        const float4 w0 = *(const float4*)(W_out + 0 * RNN_H + lane * 4);
        const float4 w1 = *(const float4*)(W_out + 1 * RNN_H + lane * 4);
        const float4 w2 = *(const float4*)(W_out + 2 * RNN_H + lane * 4);
        p0 = h.x * w0.x + h.y * w0.y + h.z * w0.z + h.w * w0.w;
        p1 = h.x * w1.x + h.y * w1.y + h.z * w1.z + h.w * w1.w;
        p2 = h.x * w2.x + h.y * w2.y + h.z * w2.z + h.w * w2.w;
    }
#pragma unroll
    for (int off = 16; off > 0; off >>= 1) {
        p0 += __shfl_xor(p0, off, 32);
        p1 += __shfl_xor(p1, off, 32);
        p2 += __shfl_xor(p2, off, 32);
    }
    if (lane == 0) {
        out[(size_t)gwave * RNN_O + 0] = p0 + b_out[0];
        out[(size_t)gwave * RNN_O + 1] = p1 + b_out[1];
        out[(size_t)gwave * RNN_O + 2] = p2 + b_out[2];
    }
}

extern "C" void kernel_launch(void* const* d_in, const int* in_sizes, int n_in,
                              void* d_out, int out_size, void* d_ws, size_t ws_size,
                              hipStream_t stream) {
    const float* inputs = (const float*)d_in[0];
    const float* W_ih   = (const float*)d_in[1];
    const float* W_hh   = (const float*)d_in[2];
    const float* b_ih   = (const float*)d_in[3];
    const float* b_hh   = (const float*)d_in[4];
    const float* h0     = (const float*)d_in[5];
    const float* W_out  = (const float*)d_in[6];
    const float* b_out  = (const float*)d_in[7];

    float* out = (float*)d_out;                                    // [B,T,O]
    float* hid = out + (size_t)RNN_B * RNN_T * RNN_O;              // [B,T,H]

    rnn_scan_kernel<<<RNN_B / MTILE, 256, 0, stream>>>(
        inputs, W_ih, W_hh, b_ih, b_hh, h0, hid);

    const int rows_per_block = 256 / 32; // 8 waves -> 8 (b,t) rows
    out_proj_kernel<<<(RNN_B * RNN_T) / rows_per_block, 256, 0, stream>>>(
        hid, W_out, b_out, out);
}
